// Model_55963423867312
// MI455X (gfx1250) — compile-verified, matrix operand order checked by reading, not run on hardware
//
#include <hip/hip_runtime.h>
#include <cstdint>
#include <cstddef>

// ---------------------------------------------------------------------------
// MI455X (gfx1250) implementation of the neural-HMM aligner forward pass.
// All GEMMs run through v_wmma_f32_16x16x32_bf16 (wave32, f32 accum).
// ---------------------------------------------------------------------------

typedef __bf16 bf16;
typedef __attribute__((ext_vector_type(8)))  __bf16 v8bf;
typedef __attribute__((ext_vector_type(16))) __bf16 v16bf;
typedef __attribute__((ext_vector_type(8)))  float  v8f;

#define DEV __device__ __forceinline__

// Load one 16-lane-pattern bf16 fragment: elements [0..7] = K base..base+7,
// elements [8..15] = K base+16..base+23 (ISA 16-bit A/B striping; the caller
// folds the lane's hi/lo half offset (hi8) and row*ld into `p`).
DEV v16bf ldfrag(const bf16* p) {
  v8bf lo = *(const v8bf*)(p);
  v8bf hi = *(const v8bf*)(p + 16);
  return __builtin_shufflevector(lo, hi, 0,1,2,3,4,5,6,7,8,9,10,11,12,13,14,15);
}

DEV v8f wmma_bf(v16bf a, v16bf b, v8f c) {
  return __builtin_amdgcn_wmma_f32_16x16x32_bf16(false, a, false, b, (short)0, c,
                                                 false, false);
}

DEV float sigf(float x) { return 1.f / (1.f + expf(-x)); }

// Problem constants
#define BB 16
#define JJ 96
#define II 96
#define EE 256
#define HH 512
#define VV 32000
#define SJ 193
#define SJP 208   // SJ padded to 16

// ---------------------------------------------------------------------------
// LDS-tiled transpose + f32->bf16:  dst[n][k] = src[k][n]; zero pad n>=C.
// src is R x C (f32), dst is dstRows x R (bf16). block (32,8).
// ---------------------------------------------------------------------------
__global__ void k_transpose(const float* __restrict__ src, bf16* __restrict__ dst,
                            int R, int C, int dstRows) {
  __shared__ float tile[32][33];
  int kt = blockIdx.y * 32;   // k block (src row)
  int nt = blockIdx.x * 32;   // n block (src col)
  int tx = threadIdx.x, ty = threadIdx.y;
  for (int i = ty; i < 32; i += 8) {
    int k = kt + i, n = nt + tx;
    tile[i][tx] = (k < R && n < C) ? src[(size_t)k * C + n] : 0.f;
  }
  __syncthreads();
  for (int i = ty; i < 32; i += 8) {
    int n = nt + i, k = kt + tx;
    if (n < dstRows && k < R)
      dst[(size_t)n * R + k] = (bf16)tile[tx][i];
  }
}

// e_word gather: eW[b*96+j][e] = bf16(emb[targets[b*96+j]][e])
__global__ void k_gather_emb(const int* __restrict__ tgt, const float* __restrict__ emb,
                             bf16* __restrict__ ew) {
  int r = blockIdx.x, e = threadIdx.x;
  ew[(size_t)r * EE + e] = (bf16)emb[(size_t)tgt[r] * EE + e];
}

// tn = tanh(e_null @ W1 + b1); p0 = sigmoid(tn . Wp + bp) * 0.3
__global__ void k_tn(const int* __restrict__ tnull, const float* __restrict__ emb,
                     const float* __restrict__ W1, const float* __restrict__ b1,
                     const float* __restrict__ Wp, const float* __restrict__ bp,
                     float* __restrict__ tn, float* __restrict__ p0) {
  __shared__ float red[512];
  int n = threadIdx.x;
  const float* e = emb + (size_t)tnull[0] * EE;
  float acc = b1[n];
  for (int k = 0; k < EE; ++k) acc += e[k] * W1[(size_t)k * HH + n];
  float t = tanhf(acc);
  tn[n] = t;
  red[n] = t * Wp[n];
  __syncthreads();
  for (int s = 256; s > 0; s >>= 1) {
    if (n < s) red[n] += red[n + s];
    __syncthreads();
  }
  if (n == 0) p0[0] = sigf(red[0] + bp[0]) * 0.3f;
}

// X-gate precompute: Xg[dir][j][b][n] = e_word[b][j] @ Wx_dir + b_dir
// grid (16, 96, 2), block 256 (8 waves); one wave = one 16x16 tile.
__global__ void k_xgate(const bf16* __restrict__ ew,
                        const bf16* __restrict__ wfwT, const bf16* __restrict__ wbwT,
                        const float* __restrict__ bfw, const float* __restrict__ bbw,
                        float* __restrict__ xgf, float* __restrict__ xgb) {
  int w = threadIdx.x >> 5, lane = threadIdx.x & 31;
  int nl = lane & 15, hi8 = (lane >> 4) << 3, mb = (lane >> 4) << 3;
  int ntile = blockIdx.x * 8 + w;
  int j = blockIdx.y;
  int dir = blockIdx.z;
  const bf16* WT   = dir ? wbwT : wfwT;
  const float* bia = dir ? bbw : bfw;
  float* Xg        = dir ? xgb : xgf;
  const bf16* arow = ew + ((size_t)nl * JJ + j) * EE;       // A row = batch nl
  int n = ntile * 16 + nl;
  const bf16* brow = WT + (size_t)n * 768;                  // WT: [2048][768]
  v8f acc = {};
#pragma unroll
  for (int kt = 0; kt < 8; ++kt) {
    int kb = kt * 32 + hi8;
    acc = wmma_bf(ldfrag(arow + kb), ldfrag(brow + kb), acc);
  }
  float bn = bia[n];
#pragma unroll
  for (int r = 0; r < 8; ++r)
    Xg[((size_t)j * BB + (mb + r)) * 2048 + n] = acc[r] + bn;
}

// BiLSTM recurrence: grid 2 (dir), block 512 (16 waves). Wave w owns hidden
// columns [w*32, w*32+32) across all 4 gates; c lives in VGPRs, h in LDS bf16.
__global__ void k_lstm(const float* __restrict__ xgf, const float* __restrict__ xgb,
                       const bf16* __restrict__ wfwT, const bf16* __restrict__ wbwT,
                       bf16* __restrict__ hcat) {
  __shared__ __align__(16) bf16 hlds[BB * HH];
  int dir = blockIdx.x;
  const float* Xg = dir ? xgb : xgf;
  const bf16* WT  = dir ? wbwT : wfwT;
  int dcol = dir ? HH : 0;
  int tid = threadIdx.x, w = tid >> 5, lane = tid & 31;
  int nl = lane & 15, hi8 = (lane >> 4) << 3, mb = (lane >> 4) << 3;
  v8f cst[2];
  cst[0] = {}; cst[1] = {};
  for (int t = 0; t < JJ; ++t) {
    int j = dir ? (JJ - 1 - t) : t;
    v8f acc[4][2];
#pragma unroll
    for (int g = 0; g < 4; ++g)
#pragma unroll
      for (int s = 0; s < 2; ++s) {
        int n0 = g * HH + w * 32 + s * 16 + nl;
#pragma unroll
        for (int r = 0; r < 8; ++r)
          acc[g][s][r] = Xg[((size_t)j * BB + (mb + r)) * 2048 + n0];
      }
    if (t > 0) {
      for (int kt = 0; kt < 16; ++kt) {
        int kb = kt * 32 + hi8;
        v16bf a = ldfrag(hlds + nl * HH + kb);
#pragma unroll
        for (int g = 0; g < 4; ++g)
#pragma unroll
          for (int s = 0; s < 2; ++s) {
            int n = g * HH + w * 32 + s * 16 + nl;
            v16bf b = ldfrag(WT + (size_t)n * 768 + EE + kb);  // h-part K in [256,768)
            acc[g][s] = wmma_bf(a, b, acc[g][s]);
          }
      }
    }
    __syncthreads();  // all reads of hlds done
#pragma unroll
    for (int s = 0; s < 2; ++s) {
#pragma unroll
      for (int r = 0; r < 8; ++r) {
        float zi = acc[0][s][r], zg = acc[1][s][r], zf = acc[2][s][r], zo = acc[3][s][r];
        float cc = sigf(zf + 1.f) * cst[s][r] + sigf(zi) * tanhf(zg);
        cst[s][r] = cc;
        float h = sigf(zo) * tanhf(cc);
        int m = mb + r, hc = w * 32 + s * 16 + nl;
        bf16 hb = (bf16)h;
        hlds[m * HH + hc] = hb;
        hcat[((size_t)m * JJ + j) * (2 * HH) + dcol + hc] = hb;
      }
    }
    __syncthreads();  // new h visible before next step
  }
}

// tsE = hcat @ W_lstm    [1536,1024]x[1024,256] -> bf16
__global__ void k_ts1(const bf16* __restrict__ hcat, const bf16* __restrict__ wlT,
                      bf16* __restrict__ tse) {
  int w = threadIdx.x >> 5, lane = threadIdx.x & 31;
  int nl = lane & 15, hi8 = (lane >> 4) << 3, mb = (lane >> 4) << 3;
  int ntile = blockIdx.x * 8 + w, mtile = blockIdx.y;
  const bf16* arow = hcat + (size_t)(mtile * 16 + nl) * 1024;
  const bf16* brow = wlT + (size_t)(ntile * 16 + nl) * 1024;
  v8f acc = {};
  for (int kt = 0; kt < 32; ++kt) {
    int kb = kt * 32 + hi8;
    acc = wmma_bf(ldfrag(arow + kb), ldfrag(brow + kb), acc);
  }
#pragma unroll
  for (int r = 0; r < 8; ++r)
    tse[(size_t)(mtile * 16 + mb + r) * EE + ntile * 16 + nl] = (bf16)acc[r];
}

// ts = tanh(tsE @ W1 + b1)  [1536,256]x[256,512] -> bf16
__global__ void k_ts2(const bf16* __restrict__ tse, const bf16* __restrict__ w1T,
                      const float* __restrict__ b1, bf16* __restrict__ ts) {
  int w = threadIdx.x >> 5, lane = threadIdx.x & 31;
  int nl = lane & 15, hi8 = (lane >> 4) << 3, mb = (lane >> 4) << 3;
  int ntile = blockIdx.x * 8 + w, mtile = blockIdx.y;
  const bf16* arow = tse + (size_t)(mtile * 16 + nl) * EE;
  int n = ntile * 16 + nl;
  const bf16* brow = w1T + (size_t)n * EE;
  v8f acc = {};
#pragma unroll
  for (int kt = 0; kt < 8; ++kt) {
    int kb = kt * 32 + hi8;
    acc = wmma_bf(ldfrag(arow + kb), ldfrag(brow + kb), acc);
  }
  float bn = b1[n];
#pragma unroll
  for (int r = 0; r < 8; ++r)
    ts[(size_t)(mtile * 16 + mb + r) * HH + n] = (bf16)tanhf(acc[r] + bn);
}

// Streaming softmax stats over V=32000: per-row (max, sumexp) of ts@Wv+bv.
// grid 96 (row tile), block 256 (8 waves, each sweeps 250 n-tiles).
// A fragments (16 KB of ts rows) are hoisted into registers across the sweep;
// B stream gets a one-tile-ahead global_prefetch.
__global__ void k_vstats(const bf16* __restrict__ ts, const bf16* __restrict__ wvT,
                         const float* __restrict__ bv,
                         float* __restrict__ rowmax, float* __restrict__ rowsum) {
  int mtile = blockIdx.x;
  int tid = threadIdx.x, w = tid >> 5, lane = tid & 31;
  int nl = lane & 15, hi8 = (lane >> 4) << 3, mb = (lane >> 4) << 3;
  const bf16* arow = ts + (size_t)(mtile * 16 + nl) * HH;
  v16bf af[16];
#pragma unroll
  for (int kt = 0; kt < 16; ++kt) af[kt] = ldfrag(arow + kt * 32 + hi8);
  float mx[8], sm[8];
#pragma unroll
  for (int r = 0; r < 8; ++r) { mx[r] = -3.0e38f; sm[r] = 0.f; }
  for (int nt = w; nt < VV / 16; nt += 8) {
    const bf16* brow = wvT + (size_t)(nt * 16 + nl) * HH;
    if (nt + 8 < VV / 16)
      __builtin_prefetch(brow + (size_t)8 * 16 * HH, 0, 1);  // next tile for this wave
    v8f acc = {};
#pragma unroll
    for (int kt = 0; kt < 16; ++kt)
      acc = wmma_bf(af[kt], ldfrag(brow + kt * 32 + hi8), acc);
    float bn = bv[nt * 16 + nl];
#pragma unroll
    for (int r = 0; r < 8; ++r) {
      float z = acc[r] + bn;
      float mn = fmaxf(mx[r], z);
      sm[r] = sm[r] * expf(mx[r] - mn) + expf(z - mn);
      mx[r] = mn;
    }
  }
  // butterfly merge over the 16 lanes sharing the same rows (stays in half)
#pragma unroll
  for (int off = 1; off < 16; off <<= 1) {
#pragma unroll
    for (int r = 0; r < 8; ++r) {
      float om = __shfl_xor(mx[r], off, 32);
      float os = __shfl_xor(sm[r], off, 32);
      float mn = fmaxf(mx[r], om);
      sm[r] = sm[r] * expf(mx[r] - mn) + os * expf(om - mn);
      mx[r] = mn;
    }
  }
  __shared__ float Lm[16][8], Ls[16][8];
  if (nl == 0) {
#pragma unroll
    for (int r = 0; r < 8; ++r) { Lm[mb + r][w] = mx[r]; Ls[mb + r][w] = sm[r]; }
  }
  __syncthreads();
  if (tid < 16) {
    float M = -3.0e38f, S = 0.f;
    for (int ww = 0; ww < 8; ++ww) {
      float m2 = Lm[tid][ww], s2 = Ls[tid][ww];
      float mn = fmaxf(M, m2);
      S = S * expf(M - mn) + s2 * expf(m2 - mn);
      M = mn;
    }
    rowmax[mtile * 16 + tid] = M;
    rowsum[mtile * 16 + tid] = S;
  }
}

// Gathered emission GEMM: em_word[b][j][i] = softmax_row(ts[b,j]@Wv+bv)[src[b,i]].
// grid (6 ntile, 6 mtile, 16 b), block 32 (one wave per tile).
__global__ void k_emword(const bf16* __restrict__ ts, const bf16* __restrict__ wvT,
                         const float* __restrict__ bv, const int* __restrict__ src,
                         const float* __restrict__ rowmax, const float* __restrict__ rowsum,
                         float* __restrict__ out) {
  int b = blockIdx.z, mtile = blockIdx.y, ntile = blockIdx.x;
  int lane = threadIdx.x & 31;
  int nl = lane & 15, hi8 = (lane >> 4) << 3, mb = (lane >> 4) << 3;
  const bf16* arow = ts + (size_t)(b * JJ + mtile * 16 + nl) * HH;
  int i = ntile * 16 + nl;
  int sv = src[b * II + i];
  const bf16* brow = wvT + (size_t)sv * HH;
  v8f acc = {};
  for (int kt = 0; kt < 16; ++kt) {
    int kb = kt * 32 + hi8;
    acc = wmma_bf(ldfrag(arow + kb), ldfrag(brow + kb), acc);
  }
  float bn = bv[sv];
#pragma unroll
  for (int r = 0; r < 8; ++r) {
    int j = mtile * 16 + mb + r;
    int row = b * JJ + j;
    float v = expf(acc[r] + bn - rowmax[row]) / rowsum[row];
    out[(size_t)b * (2 * JJ) * II + (size_t)j * II + i] = v;
  }
}

// Null-row softmax stats (single row over V), stage 1: per-block partials.
__global__ void k_nullstats1(const float* __restrict__ tn, const float* __restrict__ Wv,
                             const float* __restrict__ bv, float* __restrict__ part) {
  __shared__ float red[256];
  int tid = threadIdx.x;
  int v = blockIdx.x * 256 + tid;     // 125*256 == 32000 exactly
  float acc = bv[v];
  for (int k = 0; k < HH; ++k) acc += tn[k] * Wv[(size_t)k * VV + v];
  red[tid] = acc;
  __syncthreads();
  for (int s = 128; s > 0; s >>= 1) {
    if (tid < s) red[tid] = fmaxf(red[tid], red[tid + s]);
    __syncthreads();
  }
  float bmax = red[0];
  __syncthreads();
  red[tid] = expf(acc - bmax);
  __syncthreads();
  for (int s = 128; s > 0; s >>= 1) {
    if (tid < s) red[tid] += red[tid + s];
    __syncthreads();
  }
  if (tid == 0) { part[blockIdx.x * 2] = bmax; part[blockIdx.x * 2 + 1] = red[0]; }
}

__global__ void k_nullstats2(const float* __restrict__ part, float* __restrict__ nstat) {
  if (threadIdx.x == 0) {
    float M = -3.0e38f, S = 0.f;
    for (int i = 0; i < 125; ++i) {
      float m2 = part[2 * i], s2 = part[2 * i + 1];
      float mn = fmaxf(M, m2);
      S = S * expf(M - mn) + s2 * expf(m2 - mn);
      M = mn;
    }
    nstat[0] = M; nstat[1] = S;
  }
}

// em_null[b][i] = p_null[src[b,i]] broadcast across rows 96..191 of emission.
__global__ void k_emnull(const float* __restrict__ tn, const float* __restrict__ Wv,
                         const float* __restrict__ bv, const int* __restrict__ src,
                         const float* __restrict__ nstat, float* __restrict__ out) {
  int b = blockIdx.x, i = threadIdx.x;   // block 96
  int sv = src[b * II + i];
  float acc = bv[sv];
  for (int k = 0; k < HH; ++k) acc += tn[k] * Wv[(size_t)k * VV + sv];
  float v = expf(acc - nstat[0]) / nstat[1];
  float* o = out + (size_t)b * (2 * JJ) * II;
  for (int j = 0; j < JJ; ++j) o[(size_t)(JJ + j) * II + i] = v;
}

// jw = softmax(ts @ Wj + bj) over 193 cols. grid 96 (mtile), block 32 (1 wave).
__global__ void k_jw(const bf16* __restrict__ ts, const bf16* __restrict__ wjT,
                     const float* __restrict__ bj, float* __restrict__ jw) {
  int mtile = blockIdx.x;
  int lane = threadIdx.x & 31;
  int nl = lane & 15, hi8 = (lane >> 4) << 3, mb = (lane >> 4) << 3;
  const bf16* arow = ts + (size_t)(mtile * 16 + nl) * HH;
  float z[13][8];
  for (int nt = 0; nt < 13; ++nt) {
    const bf16* brow = wjT + (size_t)(nt * 16 + nl) * HH;
    v8f acc = {};
    for (int kt = 0; kt < 16; ++kt) {
      int kb = kt * 32 + hi8;
      acc = wmma_bf(ldfrag(arow + kb), ldfrag(brow + kb), acc);
    }
    int n = nt * 16 + nl;
    float bn = (n < SJ) ? bj[n] : 0.f;
#pragma unroll
    for (int r = 0; r < 8; ++r) z[nt][r] = acc[r] + bn;
  }
  float mx[8], sm[8];
#pragma unroll
  for (int r = 0; r < 8; ++r) {
    float m = -3.0e38f;
    for (int nt = 0; nt < 13; ++nt)
      if (nt * 16 + nl < SJ) m = fmaxf(m, z[nt][r]);
    mx[r] = m;
  }
#pragma unroll
  for (int off = 1; off < 16; off <<= 1)
#pragma unroll
    for (int r = 0; r < 8; ++r) mx[r] = fmaxf(mx[r], __shfl_xor(mx[r], off, 32));
#pragma unroll
  for (int r = 0; r < 8; ++r) {
    float s = 0.f;
    for (int nt = 0; nt < 13; ++nt)
      if (nt * 16 + nl < SJ) s += expf(z[nt][r] - mx[r]);
    sm[r] = s;
  }
#pragma unroll
  for (int off = 1; off < 16; off <<= 1)
#pragma unroll
    for (int r = 0; r < 8; ++r) sm[r] += __shfl_xor(sm[r], off, 32);
  for (int nt = 0; nt < 13; ++nt) {
    int n = nt * 16 + nl;
    if (n < SJ)
#pragma unroll
      for (int r = 0; r < 8; ++r)
        jw[(size_t)(mtile * 16 + mb + r) * SJ + n] = expf(z[nt][r] - mx[r]) / sm[r];
  }
}

// transition + transition_log from jw and p0. grid (96 j1, 16 b), block 96 (j2).
__global__ void k_trans(const float* __restrict__ jw, const float* __restrict__ p0p,
                        float* __restrict__ out) {
  int j1 = blockIdx.x, b = blockIdx.y, j2 = threadIdx.x;
  __shared__ float s[96];
  __shared__ float tot;
  float v = jw[(size_t)(b * JJ + j1) * SJ + (96 - j1 + j2)];
  s[j2] = v;
  __syncthreads();
  if (j2 == 0) {
    float t = 0.f;
    for (int i = 0; i < JJ; ++i) t += s[i];
    tot = t;
  }
  __syncthreads();
  float p0 = p0p[0];
  float jn = v * (1.f - p0) / tot;
  float lj = logf(jn);
  float eye = (j1 == j2) ? p0 : 0.f;
  float leye = (j1 == j2) ? logf(p0) : 0.f;
  size_t Tb = (size_t)294912 + (size_t)b * 36864;
  size_t Lb = (size_t)884736 + (size_t)b * 36864;
  out[Tb + (size_t)j1 * 192 + j2] = jn;
  out[Tb + (size_t)(j1 + 96) * 192 + j2] = jn;
  out[Tb + (size_t)j1 * 192 + 96 + j2] = eye;
  out[Tb + (size_t)(j1 + 96) * 192 + 96 + j2] = eye;
  out[Lb + (size_t)j1 * 192 + j2] = lj;
  out[Lb + (size_t)(j1 + 96) * 192 + j2] = lj;
  out[Lb + (size_t)j1 * 192 + 96 + j2] = leye;
  out[Lb + (size_t)(j1 + 96) * 192 + 96 + j2] = leye;
}

// ---------------------------------------------------------------------------
extern "C" void kernel_launch(void* const* d_in, const int* in_sizes, int n_in,
                              void* d_out, int out_size, void* d_ws, size_t ws_size,
                              hipStream_t stream) {
  (void)in_sizes; (void)n_in; (void)out_size;
  const int*   sources = (const int*)d_in[0];
  const int*   targets = (const int*)d_in[1];
  const int*   tnull   = (const int*)d_in[2];
  const float* emb     = (const float*)d_in[3];
  const float* W_fw    = (const float*)d_in[4];
  const float* b_fw    = (const float*)d_in[5];
  const float* W_bw    = (const float*)d_in[6];
  const float* b_bw    = (const float*)d_in[7];
  const float* W_lstm  = (const float*)d_in[8];
  const float* W1      = (const float*)d_in[9];
  const float* b1      = (const float*)d_in[10];
  const float* Wv      = (const float*)d_in[11];
  const float* bv      = (const float*)d_in[12];
  const float* Wj      = (const float*)d_in[13];
  const float* bj      = (const float*)d_in[14];
  const float* Wp      = (const float*)d_in[15];
  const float* bp      = (const float*)d_in[16];
  float* out = (float*)d_out;

  // bump allocator over scratch (deterministic)
  char* p = (char*)d_ws;
  auto alloc = [&](size_t bytes) -> void* {
    void* r = (void*)p;
    p += (bytes + 255) & ~(size_t)255;
    return r;
  };
  bf16* eW    = (bf16*)alloc((size_t)1536 * 256 * 2);
  bf16* WfwT  = (bf16*)alloc((size_t)2048 * 768 * 2);
  bf16* WbwT  = (bf16*)alloc((size_t)2048 * 768 * 2);
  bf16* WlT   = (bf16*)alloc((size_t)256 * 1024 * 2);
  bf16* W1T   = (bf16*)alloc((size_t)512 * 256 * 2);
  bf16* WvT   = (bf16*)alloc((size_t)VV * 512 * 2);
  bf16* WjT   = (bf16*)alloc((size_t)SJP * 512 * 2);
  float* Xgf  = (float*)alloc((size_t)96 * 16 * 2048 * 4);
  float* Xgb  = (float*)alloc((size_t)96 * 16 * 2048 * 4);
  bf16* hcat  = (bf16*)alloc((size_t)1536 * 1024 * 2);
  bf16* tse   = (bf16*)alloc((size_t)1536 * 256 * 2);
  bf16* tsb   = (bf16*)alloc((size_t)1536 * 512 * 2);
  float* tn   = (float*)alloc(512 * 4);
  float* p0   = (float*)alloc(256);
  float* rmax = (float*)alloc(1536 * 4);
  float* rsum = (float*)alloc(1536 * 4);
  float* part = (float*)alloc(125 * 2 * 4);
  float* nst  = (float*)alloc(256);
  float* jw   = (float*)alloc((size_t)1536 * SJ * 4);
  if ((size_t)(p - (char*)d_ws) > ws_size) return;  // scratch too small

  dim3 tb(32, 8);
  // weight transposes -> bf16 [N][K]
  k_transpose<<<dim3(64, 24), tb, 0, stream>>>(W_fw, WfwT, 768, 2048, 2048);
  k_transpose<<<dim3(64, 24), tb, 0, stream>>>(W_bw, WbwT, 768, 2048, 2048);
  k_transpose<<<dim3(8, 32), tb, 0, stream>>>(W_lstm, WlT, 1024, 256, 256);
  k_transpose<<<dim3(16, 8), tb, 0, stream>>>(W1, W1T, 256, 512, 512);
  k_transpose<<<dim3(1000, 16), tb, 0, stream>>>(Wv, WvT, 512, VV, VV);
  k_transpose<<<dim3(7, 16), tb, 0, stream>>>(Wj, WjT, 512, SJ, SJP);

  k_gather_emb<<<1536, 256, 0, stream>>>(targets, emb, eW);
  k_tn<<<1, 512, 0, stream>>>(tnull, emb, W1, b1, Wp, bp, tn, p0);

  k_xgate<<<dim3(16, 96, 2), 256, 0, stream>>>(eW, WfwT, WbwT, b_fw, b_bw, Xgf, Xgb);
  k_lstm<<<2, 512, 0, stream>>>(Xgf, Xgb, WfwT, WbwT, hcat);

  k_ts1<<<dim3(2, 96), 256, 0, stream>>>(hcat, WlT, tse);
  k_ts2<<<dim3(4, 96), 256, 0, stream>>>(tse, W1T, b1, tsb);

  k_vstats<<<96, 256, 0, stream>>>(tsb, WvT, bv, rmax, rsum);
  k_emword<<<dim3(6, 6, 16), 32, 0, stream>>>(tsb, WvT, bv, sources, rmax, rsum, out);

  k_nullstats1<<<125, 256, 0, stream>>>(tn, Wv, bv, part);
  k_nullstats2<<<1, 32, 0, stream>>>(part, nst);
  k_emnull<<<16, 96, 0, stream>>>(tn, Wv, bv, sources, nst, out);

  k_jw<<<96, 32, 0, stream>>>(tsb, WjT, bj, jw);
  k_trans<<<dim3(96, 16), 96, 0, stream>>>(jw, p0, out);
}
